// CrossGCF_24343874633751
// MI455X (gfx1250) — compile-verified, matrix operand order checked by reading, not run on hardware
//
#include <hip/hip_runtime.h>

// ---------------------------------------------------------------------------
// CDNA5 (gfx1250) implementation of the CrossGCF layer.
// Heavy edge/node GEMMs run on v_wmma_f32_16x16x32_bf16 (wave32).
// ---------------------------------------------------------------------------

typedef __attribute__((ext_vector_type(16))) __bf16 v16bf;
typedef __attribute__((ext_vector_type(2)))  __bf16 v2bf;
typedef __attribute__((ext_vector_type(8)))  float  v8f;

#define D128 128

union FragBF {
  v16bf v;
  uint4 q[2];
};

// round-to-nearest-even f32 -> bf16 (manual fallback)
__device__ __forceinline__ unsigned short f2bf(float x) {
  unsigned u = __float_as_uint(x);
  u += 0x7FFFu + ((u >> 16) & 1u);
  return (unsigned short)(u >> 16);
}

#if defined(__has_builtin)
#if __has_builtin(__builtin_amdgcn_cvt_pk_bf16_f32)
#define HAVE_PK_BF16 1
#endif
#endif

// pack two f32 into two bf16 in one dword
__device__ __forceinline__ unsigned pk2bf(float a, float b) {
#ifdef HAVE_PK_BF16
  v2bf r = __builtin_amdgcn_cvt_pk_bf16_f32(a, b);
  return __builtin_bit_cast(unsigned, r);
#else
  // gfx1250 has native bf16 ALU: let the backend legalize fptrunc (ideally to
  // a packed HW convert); clang inlines this (no libcalls on amdgcn).
  v2bf r = {(__bf16)a, (__bf16)b};
  return __builtin_bit_cast(unsigned, r);
#endif
}

// order-preserving float <-> uint map for atomicMax-based segment max
__device__ __forceinline__ unsigned fkey(float f) {
  unsigned u = __float_as_uint(f);
  return (u & 0x80000000u) ? ~u : (u | 0x80000000u);
}
__device__ __forceinline__ float fkeyinv(unsigned k) {
  unsigned u = (k & 0x80000000u) ? (k & 0x7FFFFFFFu) : ~k;
  return __uint_as_float(u);
}

// ---------------------------------------------------------------------------
// K0: f32 weights -> bf16 (row-major [DO][D], i.e. already B = W^T layout by n,k)
// ---------------------------------------------------------------------------
__global__ __launch_bounds__(256) void conv_w_kernel(
    const float* __restrict__ W1, const float* __restrict__ W2,
    unsigned short* __restrict__ W1bf, unsigned short* __restrict__ W2bf) {
  int i = blockIdx.x * blockDim.x + threadIdx.x;
  if (i < D128 * D128) {
    W1bf[i] = f2bf(W1[i]);
    W2bf[i] = f2bf(W2[i]);
  }
}

// ---------------------------------------------------------------------------
// K1: self term  out[r, :] = feat[r, :] @ W1^T + b1   (full overwrite of out)
// 8 waves per block; wave w owns output columns [16w, 16w+16).
// ---------------------------------------------------------------------------
__global__ __launch_bounds__(256) void node_linear_kernel(
    const float* __restrict__ feat, const unsigned short* __restrict__ W1bf,
    const float* __restrict__ b1, float* __restrict__ out, int nTiles) {
  __shared__ __align__(16) unsigned short As[16 * D128];

  const int t     = threadIdx.x;
  const int lane  = t & 31;
  const int wave  = t >> 5;
  const int nn    = wave * 16 + (lane & 15);          // output column
  const int khalf = (lane & 16) ? 8 : 0;              // ISA 16-bit operand layout

  FragBF B1[4];
#pragma unroll
  for (int kb = 0; kb < 4; ++kb) {
    const unsigned short* p = W1bf + nn * D128 + kb * 32 + khalf;
    B1[kb].q[0] = *(const uint4*)p;
    B1[kb].q[1] = *(const uint4*)(p + 16);
  }
  const float b1n = b1[nn];

  const int el = t >> 4;          // row within 16-row tile
  const int k0 = (t & 15) * 8;    // 8 features per thread

  for (int tile = blockIdx.x; tile < nTiles; tile += gridDim.x) {
    const int rBase = tile * 16;
    const float* ps = feat + (size_t)(rBase + el) * D128 + k0;
    float4 s0 = *(const float4*)ps;
    float4 s1 = *(const float4*)(ps + 4);
    uint4 us;
    us.x = pk2bf(s0.x, s0.y);
    us.y = pk2bf(s0.z, s0.w);
    us.z = pk2bf(s1.x, s1.y);
    us.w = pk2bf(s1.z, s1.w);
    *(uint4*)&As[el * D128 + k0] = us;
    __syncthreads();

    // preload all 4 A fragments (8 ds_load_b128) so LDS latency overlaps WMMA
    const unsigned short* rowp = &As[(lane & 15) * D128];
    FragBF fa[4];
#pragma unroll
    for (int kb = 0; kb < 4; ++kb) {
      const unsigned short* pa = rowp + kb * 32 + khalf;
      fa[kb].q[0] = *(const uint4*)pa;
      fa[kb].q[1] = *(const uint4*)(pa + 16);
    }
    v8f acc = {};
#pragma unroll
    for (int kb = 0; kb < 4; ++kb) {
      acc = __builtin_amdgcn_wmma_f32_16x16x32_bf16(false, fa[kb].v, false, B1[kb].v,
                                                    (short)0, acc, false, false);
    }

    // rows are contiguous: one base pointer, constant 512B-stride stores
    const int moff = (lane & 16) ? 8 : 0;
    float* pout = out + (size_t)(rBase + moff) * D128 + nn;
#pragma unroll
    for (int v = 0; v < 8; ++v) {
      pout[v * D128] = acc[v] + b1n;
    }
    __syncthreads();
  }
}

// ---------------------------------------------------------------------------
// K2: attention logits  a_e = leaky(xs . aw[:128] + xd . aw[128:])  (wave/edge)
// plus segment-max over dst via order-preserving uint atomicMax.
// ---------------------------------------------------------------------------
__global__ __launch_bounds__(256) void attn_logits_kernel(
    const float* __restrict__ featS, const float* __restrict__ featD,
    const int* __restrict__ srcI, const int* __restrict__ dstI,
    const float* __restrict__ attn, float* __restrict__ aE,
    unsigned* __restrict__ amax, int E) {
  const int lane = threadIdx.x & 31;
  const int wave = (blockIdx.x * (blockDim.x >> 5)) + (threadIdx.x >> 5);
  const int nW   = gridDim.x * (blockDim.x >> 5);

  const float4 awS = *(const float4*)(attn + lane * 4);
  const float4 awD = *(const float4*)(attn + D128 + lane * 4);

  for (int e = wave; e < E; e += nW) {
    const int s = srcI[e];
    const int d = dstI[e];
    const float4 xs = *(const float4*)(featS + (size_t)s * D128 + lane * 4);
    const float4 xd = *(const float4*)(featD + (size_t)d * D128 + lane * 4);
    float acc = xs.x * awS.x + xs.y * awS.y + xs.z * awS.z + xs.w * awS.w
              + xd.x * awD.x + xd.y * awD.y + xd.z * awD.z + xd.w * awD.w;
#pragma unroll
    for (int o = 16; o > 0; o >>= 1) acc += __shfl_xor(acc, o, 32);
    if (lane == 0) {
      const float a = (acc >= 0.0f) ? acc : 0.2f * acc;
      aE[e] = a;
      atomicMax(amax + d, fkey(a));
    }
  }
}

// ---------------------------------------------------------------------------
// K3: ex = exp(a - amax[dst]); denom[dst] += ex   (thread/edge)
// ---------------------------------------------------------------------------
__global__ __launch_bounds__(256) void attn_norm_kernel(
    const int* __restrict__ dstI, float* __restrict__ aE,
    const unsigned* __restrict__ amax, float* __restrict__ denom, int E) {
  int e = blockIdx.x * blockDim.x + threadIdx.x;
  if (e < E) {
    const int d = dstI[e];
    const float ex = __expf(aE[e] - fkeyinv(amax[d]));
    aE[e] = ex;
    atomicAdd(denom + d, ex);
  }
}

// ---------------------------------------------------------------------------
// K4: main edge pass.  64 edges (four 16-row A-subtiles) per iteration:
//   msg = xs@W1^T + (xs*xd)@W2^T + (b1+b2)    (32 chained bf16 WMMAs / wave)
//   outD[dst] += (alpha * norm) * msg         (f32 global atomics)
// B fragments (W1,W2 x 4 K-blocks) are register-resident and reused by all
// four subtiles; per-edge output row pointers are staged in LDS so the
// scatter loop is pure atomic adds with no 64-bit index math.
// ---------------------------------------------------------------------------
__global__ __launch_bounds__(256) void edge_msg_kernel(
    const float* __restrict__ featS, const float* __restrict__ featD,
    const int* __restrict__ srcI, const int* __restrict__ dstI,
    const float* __restrict__ normE, const float* __restrict__ exE,
    const float* __restrict__ denom,
    const unsigned short* __restrict__ W1bf, const unsigned short* __restrict__ W2bf,
    const float* __restrict__ b1, const float* __restrict__ b2,
    float* __restrict__ outD, int nTiles64) {
  __shared__ __align__(16) unsigned short As[4][16 * D128];  // bf16(xs)
  __shared__ __align__(16) unsigned short Ap[4][16 * D128];  // bf16(xs*xd)
  __shared__ float* rowP[4][16];
  __shared__ float  coefS[4][16];

  const int t     = threadIdx.x;
  const int lane  = t & 31;
  const int wave  = t >> 5;
  const int nn    = wave * 16 + (lane & 15);
  const int khalf = (lane & 16) ? 8 : 0;

  // resident B fragments: W1 and W2, 4 K-blocks each
  FragBF B1[4], B2[4];
#pragma unroll
  for (int kb = 0; kb < 4; ++kb) {
    const unsigned short* p1 = W1bf + nn * D128 + kb * 32 + khalf;
    const unsigned short* p2 = W2bf + nn * D128 + kb * 32 + khalf;
    B1[kb].q[0] = *(const uint4*)p1;
    B1[kb].q[1] = *(const uint4*)(p1 + 16);
    B2[kb].q[0] = *(const uint4*)p2;
    B2[kb].q[1] = *(const uint4*)(p2 + 16);
  }
  const float biasn = b1[nn] + b2[nn];

  const int el = t >> 4;
  const int k0 = (t & 15) * 8;

  for (int tile = blockIdx.x; tile < nTiles64; tile += gridDim.x) {
    const int eBase = tile * 64;
#pragma unroll
    for (int sub = 0; sub < 4; ++sub) {
      const int e = eBase + sub * 16 + el;
      const int s = srcI[e];
      const int d = dstI[e];
      const float* ps = featS + (size_t)s * D128 + k0;
      const float* pd = featD + (size_t)d * D128 + k0;
      float4 s0 = *(const float4*)ps;
      float4 s1 = *(const float4*)(ps + 4);
      float4 d0 = *(const float4*)pd;
      float4 d1 = *(const float4*)(pd + 4);
      uint4 us, up;
      us.x = pk2bf(s0.x, s0.y);
      us.y = pk2bf(s0.z, s0.w);
      us.z = pk2bf(s1.x, s1.y);
      us.w = pk2bf(s1.z, s1.w);
      up.x = pk2bf(s0.x * d0.x, s0.y * d0.y);
      up.y = pk2bf(s0.z * d0.z, s0.w * d0.w);
      up.z = pk2bf(s1.x * d1.x, s1.y * d1.y);
      up.w = pk2bf(s1.z * d1.z, s1.w * d1.w);
      *(uint4*)&As[sub][el * D128 + k0] = us;
      *(uint4*)&Ap[sub][el * D128 + k0] = up;
      if (k0 == 0) {
        rowP[sub][el]  = outD + (size_t)d * D128;
        coefS[sub][el] = exE[e] / denom[d] * normE[e];   // alpha * norm
      }
    }
    __syncthreads();

    const int arow = (lane & 15) * D128;
    v8f acc[4] = {{}, {}, {}, {}};
#pragma unroll
    for (int kb = 0; kb < 4; ++kb) {
      const int koff = arow + kb * 32 + khalf;
#pragma unroll
      for (int sub = 0; sub < 4; ++sub) {
        FragBF fa, fp;
        const unsigned short* pa = &As[sub][koff];
        const unsigned short* pp = &Ap[sub][koff];
        fa.q[0] = *(const uint4*)pa; fa.q[1] = *(const uint4*)(pa + 16);
        fp.q[0] = *(const uint4*)pp; fp.q[1] = *(const uint4*)(pp + 16);
        acc[sub] = __builtin_amdgcn_wmma_f32_16x16x32_bf16(
            false, fa.v, false, B1[kb].v, (short)0, acc[sub], false, false);
        acc[sub] = __builtin_amdgcn_wmma_f32_16x16x32_bf16(
            false, fp.v, false, B2[kb].v, (short)0, acc[sub], false, false);
      }
    }

    const int moff = (lane & 16) ? 8 : 0;
#pragma unroll
    for (int sub = 0; sub < 4; ++sub) {
#pragma unroll
      for (int v = 0; v < 8; ++v) {
        const int m = v + moff;
        atomicAdd(rowP[sub][m] + nn, (acc[sub][v] + biasn) * coefS[sub][m]);
      }
    }
    __syncthreads();
  }
}

// ---------------------------------------------------------------------------
// K5: h = leaky(h); h /= max(||h||_2, 1e-12)   (wave per row)
// ---------------------------------------------------------------------------
__global__ __launch_bounds__(256) void finish_kernel(float* __restrict__ out, int nRows) {
  const int lane = threadIdx.x & 31;
  const int wave = (blockIdx.x * (blockDim.x >> 5)) + (threadIdx.x >> 5);
  const int nW   = gridDim.x * (blockDim.x >> 5);

  for (int r = wave; r < nRows; r += nW) {
    float* p = out + (size_t)r * D128 + lane * 4;
    float4 x = *(const float4*)p;
    x.x = (x.x >= 0.0f) ? x.x : 0.2f * x.x;
    x.y = (x.y >= 0.0f) ? x.y : 0.2f * x.y;
    x.z = (x.z >= 0.0f) ? x.z : 0.2f * x.z;
    x.w = (x.w >= 0.0f) ? x.w : 0.2f * x.w;
    float ss = x.x * x.x + x.y * x.y + x.z * x.z + x.w * x.w;
#pragma unroll
    for (int o = 16; o > 0; o >>= 1) ss += __shfl_xor(ss, o, 32);
    const float sc = 1.0f / fmaxf(sqrtf(ss), 1e-12f);
    x.x *= sc; x.y *= sc; x.z *= sc; x.w *= sc;
    *(float4*)p = x;
  }
}

// ---------------------------------------------------------------------------
extern "C" void kernel_launch(void* const* d_in, const int* in_sizes, int n_in,
                              void* d_out, int out_size, void* d_ws, size_t ws_size,
                              hipStream_t stream) {
  // setup_inputs() order:
  const float* feat_user = (const float*)d_in[0];
  const float* feat_item = (const float*)d_in[1];
  const int*   src_u     = (const int*)d_in[2];
  const int*   dst_i     = (const int*)d_in[3];
  const float* norm_ui   = (const float*)d_in[4];
  const float* norm_iu   = (const float*)d_in[5];
  const float* W1_w      = (const float*)d_in[6];
  const float* W1_b      = (const float*)d_in[7];
  const float* W2_w      = (const float*)d_in[8];
  const float* W2_b      = (const float*)d_in[9];
  const float* attn_w    = (const float*)d_in[10];

  const int NU = in_sizes[0] / D128;
  const int NI = in_sizes[1] / D128;
  const int E  = in_sizes[2];
  const int Nmax = (NU > NI) ? NU : NI;

  float* out_user = (float*)d_out;                       // [NU,128]
  float* out_item = (float*)d_out + (size_t)NU * D128;   // [NI,128]

  // workspace layout (all 16B aligned given sizes here)
  char* ws = (char*)d_ws;
  unsigned short* W1bf = (unsigned short*)ws;                        // 32 KB
  unsigned short* W2bf = (unsigned short*)(ws + 32768);              // 32 KB
  float*    aE    = (float*)(ws + 65536);                            // E f32
  unsigned* amax  = (unsigned*)(ws + 65536 + (size_t)E * 4);         // Nmax u32
  float*    denom = (float*)(ws + 65536 + (size_t)E * 4 + (size_t)Nmax * 4);
  (void)ws_size; (void)n_in; (void)out_size;

  const int nTilesU  = NU / 16;
  const int nTilesI  = NI / 16;
  const int nTiles64 = E / 64;

  // K0: weights -> bf16
  conv_w_kernel<<<(D128 * D128 + 255) / 256, 256, 0, stream>>>(W1_w, W2_w, W1bf, W2bf);

  // K1: self terms (full overwrite of d_out)
  node_linear_kernel<<<1024, 256, 0, stream>>>(feat_user, W1bf, W1_b, out_user, nTilesU);
  node_linear_kernel<<<1024, 256, 0, stream>>>(feat_item, W1bf, W1_b, out_item, nTilesI);

  // ---- direction user -> item (dst = items) ----
  hipMemsetAsync(amax, 0, (size_t)Nmax * 4, stream);
  hipMemsetAsync(denom, 0, (size_t)Nmax * 4, stream);
  attn_logits_kernel<<<4096, 256, 0, stream>>>(feat_user, feat_item, src_u, dst_i,
                                               attn_w, aE, amax, E);
  attn_norm_kernel<<<(E + 255) / 256, 256, 0, stream>>>(dst_i, aE, amax, denom, E);
  edge_msg_kernel<<<4096, 256, 0, stream>>>(feat_user, feat_item, src_u, dst_i,
                                            norm_ui, aE, denom, W1bf, W2bf,
                                            W1_b, W2_b, out_item, nTiles64);

  // ---- direction item -> user (dst = users; edges reversed) ----
  hipMemsetAsync(amax, 0, (size_t)Nmax * 4, stream);
  hipMemsetAsync(denom, 0, (size_t)Nmax * 4, stream);
  attn_logits_kernel<<<4096, 256, 0, stream>>>(feat_item, feat_user, dst_i, src_u,
                                               attn_w, aE, amax, E);
  attn_norm_kernel<<<(E + 255) / 256, 256, 0, stream>>>(src_u, aE, amax, denom, E);
  edge_msg_kernel<<<4096, 256, 0, stream>>>(feat_item, feat_user, dst_i, src_u,
                                            norm_iu, aE, denom, W1bf, W2bf,
                                            W1_b, W2_b, out_user, nTiles64);

  // K5: leaky + L2 normalize everything
  finish_kernel<<<4096, 256, 0, stream>>>((float*)d_out, NU + NI);
}
